// Qwen3MoeExperts_6674379178220
// MI455X (gfx1250) — compile-verified, matrix operand order checked by reading, not run on hardware
//
#include <hip/hip_runtime.h>
#include <hip/hip_bf16.h>

#define E_ 8
#define H_ 2048
#define I_ 768
#define T_ 1024
#define K_ 2
#define TK 2048
#define I2 1536
#define PADROWS 2176      // TK + E_*16 padding
#define MAX_TILES 136     // PADROWS/16
#define CHUNK 128         // K-chunk staged per double-buffer slot

typedef __attribute__((ext_vector_type(16))) __bf16 v16bf;
typedef __attribute__((ext_vector_type(8)))  __bf16 v8bf;
typedef __attribute__((ext_vector_type(8)))  float  v8f;

// ---------------------------------------------------------------- utilities
__global__ void zero_out_kernel(float* __restrict__ out, int n) {
  int i = blockIdx.x * blockDim.x + threadIdx.x;
  if (i < n) out[i] = 0.f;
}

// ------------------------------------------------- routing / tile building
__global__ void route_kernel(const int* __restrict__ sel, const float* __restrict__ rw,
                             int* __restrict__ perm_token, float* __restrict__ perm_rw,
                             int* __restrict__ tile_e, int* __restrict__ tile_r0,
                             int* __restrict__ n_tiles) {
  __shared__ int cnt[E_], cur[E_], poff[E_ + 1];
  int tid = threadIdx.x;
  if (tid < E_) { cnt[tid] = 0; cur[tid] = 0; }
  __syncthreads();
  for (int i = tid; i < TK; i += blockDim.x) atomicAdd(&cnt[sel[i]], 1);
  __syncthreads();
  if (tid == 0) {
    int run = 0;
    for (int e = 0; e < E_; ++e) { poff[e] = run; run += ((cnt[e] + 15) & ~15); }
    poff[E_] = run;
  }
  __syncthreads();
  for (int p = tid; p < PADROWS; p += blockDim.x) { perm_token[p] = -1; perm_rw[p] = 0.f; }
  __syncthreads();
  for (int i = tid; i < TK; i += blockDim.x) {
    int e = sel[i];
    int r = atomicAdd(&cur[e], 1);
    int p = poff[e] + r;
    perm_token[p] = i / K_;
    perm_rw[p]    = rw[i];
  }
  __syncthreads();
  if (tid == 0) {
    int nt = 0;
    for (int e = 0; e < E_; ++e) {
      int nte = (cnt[e] + 15) >> 4;
      for (int j = 0; j < nte; ++j) { tile_e[nt] = e; tile_r0[nt] = poff[e] + 16 * j; ++nt; }
    }
    *n_tiles = nt;
  }
}

// ------------------------------------------------- permute tokens to bf16
__global__ void permute_cast_kernel(const float* __restrict__ hidden,
                                    const int* __restrict__ perm_token,
                                    __bf16* __restrict__ Xp) {
  int idx = blockIdx.x * blockDim.x + threadIdx.x;   // PADROWS * H_
  if (idx >= PADROWS * H_) return;
  int pos = idx >> 11;
  int h   = idx & (H_ - 1);
  int t = perm_token[pos];
  Xp[idx] = (t >= 0) ? (__bf16)hidden[t * H_ + h] : (__bf16)0.f;
}

// -------------------- coalesced transpose + cast: in [e][R][C] f32 -> out [e][C][R] bf16
__global__ __launch_bounds__(256)
void transpose_cast_kernel(const float* __restrict__ in, __bf16* __restrict__ out,
                           int R, int C) {
  __shared__ float tile[32][33];
  int e  = blockIdx.z;
  int c0 = blockIdx.x * 32;
  int r0 = blockIdx.y * 32;
  const float* ip = in + (size_t)e * R * C;
  __bf16* op = out + (size_t)e * R * C;
  int tx = threadIdx.x, ty = threadIdx.y;            // (32, 8)
#pragma unroll
  for (int i = 0; i < 32; i += 8)
    tile[ty + i][tx] = ip[(size_t)(r0 + ty + i) * C + c0 + tx];
  __syncthreads();
#pragma unroll
  for (int i = 0; i < 32; i += 8)
    op[(size_t)(c0 + ty + i) * R + r0 + tx] = (__bf16)tile[tx][ty + i];
}

// ---------------------------------------------------------------- WMMA frag
// bf16 operand layout (wave32): lane<16 -> row/col = lane, K {0..7,16..23};
// lane>=16 -> K {8..15,24..31}. Pointers must already include the 8*lh skew.
__device__ __forceinline__ v16bf load_frag(const __bf16* base) {
  v8bf lo = *(const v8bf*)base;
  v8bf hi = *(const v8bf*)(base + 16);
  return __builtin_shufflevector(lo, hi, 0,1,2,3,4,5,6,7,8,9,10,11,12,13,14,15);
}

#define WMMA_BF16(a, b, c) \
  __builtin_amdgcn_wmma_f32_16x16x32_bf16(false, (a), false, (b), (short)0, (c), false, false)

// async stage: each thread copies 16 bytes memory -> LDS via ASYNCcnt path
__device__ __forceinline__ void async_copy_b128(unsigned lds_addr, unsigned gl_off,
                                                unsigned long long base) {
  asm volatile("global_load_async_to_lds_b128 %0, %1, %2"
               :: "v"(lds_addr), "v"(gl_off), "s"(base) : "memory");
}
__device__ __forceinline__ void async_wait0() {
  asm volatile("s_wait_asynccnt 0" ::: "memory");
}

// --------------------------------------------- GEMM1 (fc1) + in-register SwiGLU
// block: 8 waves; wave owns 3 gate subtiles + matching 3 up subtiles (48 cols)
__global__ __launch_bounds__(256)
void gemm1_swiglu_kernel(const __bf16* __restrict__ Xp, const __bf16* __restrict__ Wg,
                         const int* __restrict__ tile_e, const int* __restrict__ tile_r0,
                         const int* __restrict__ n_tiles, __bf16* __restrict__ act) {
  if ((int)blockIdx.x >= *n_tiles) return;
  __shared__ __bf16 sA[2][16 * CHUNK];               // double buffer, 2 x 4 KB
  int tid  = threadIdx.x;
  int wave = tid >> 5;
  int lane = tid & 31;
  int lh   = lane >> 4;
  int nn   = lane & 15;
  int e    = tile_e[blockIdx.x];
  int row0 = tile_r0[blockIdx.x];
  int gcol0 = blockIdx.y * 384 + wave * 48;          // gate column base of this wave
  const __bf16* wbase = Wg + (size_t)e * I2 * H_;
  const __bf16* bG[3];
  const __bf16* bU[3];
#pragma unroll
  for (int j = 0; j < 3; ++j) {
    bG[j] = wbase + (size_t)(gcol0 + 16 * j + nn) * H_ + 8 * lh;
    bU[j] = wbase + (size_t)(I_ + gcol0 + 16 * j + nn) * H_ + 8 * lh;
  }
  v8f accG[3] = {{}, {}, {}};
  v8f accU[3] = {{}, {}, {}};

  int sm  = tid >> 4;                                // staging row 0..15
  int sc8 = (tid & 15) << 3;                         // staging col 0..120 step 8
  unsigned gl_off = (unsigned)(((row0 + sm) * H_ + sc8) * sizeof(__bf16));
  unsigned lds0 = (unsigned)(uintptr_t)&sA[0][sm * CHUNK + sc8];
  unsigned lds1 = (unsigned)(uintptr_t)&sA[1][sm * CHUNK + sc8];
  unsigned long long xbase = (unsigned long long)(uintptr_t)Xp;

  // prologue: stage chunk 0 into buffer 0
  async_copy_b128(lds0, gl_off, xbase);
  async_wait0();
  __syncthreads();

  int p = 0;
  for (int kc = 0; kc < H_; kc += CHUNK, p ^= 1) {
    if (kc + CHUNK < H_)                              // prefetch next chunk into other buffer
      async_copy_b128(p ? lds0 : lds1,
                      gl_off + (unsigned)((kc + CHUNK) * sizeof(__bf16)), xbase);
#pragma unroll
    for (int kk = 0; kk < CHUNK; kk += 32) {
      v16bf a = load_frag(&sA[p][nn * CHUNK + kk + 8 * lh]);
#pragma unroll
      for (int j = 0; j < 3; ++j) {
        v16bf b0 = load_frag(bG[j] + kc + kk);
        accG[j] = WMMA_BF16(a, b0, accG[j]);
        v16bf b1 = load_frag(bU[j] + kc + kk);
        accU[j] = WMMA_BF16(a, b1, accU[j]);
      }
    }
    async_wait0();                                    // prefetch landed (hidden behind WMMAs)
    __syncthreads();                                  // all waves done reading buf p
  }

  // SwiGLU fully in registers: gate/up accs live at identical (m,n) positions
#pragma unroll
  for (int j = 0; j < 3; ++j) {
    int col = gcol0 + 16 * j + nn;
#pragma unroll
    for (int r = 0; r < 8; ++r) {
      int m = r + 8 * lh;
      float g = accG[j][r];
      float u = accU[j][r];
      float s = g / (1.f + __expf(-g));
      act[(size_t)(row0 + m) * I_ + col] = (__bf16)(s * u);
    }
  }
}

// ------------------------------------- GEMM2 (fc2) + weighted atomic combine
// block: 8 waves; wave owns 4 subtiles (64 hidden cols)
__global__ __launch_bounds__(256)
void gemm2_combine_kernel(const __bf16* __restrict__ act, const __bf16* __restrict__ Wd,
                          const int* __restrict__ tile_e, const int* __restrict__ tile_r0,
                          const int* __restrict__ n_tiles,
                          const int* __restrict__ perm_token, const float* __restrict__ perm_rw,
                          float* __restrict__ out) {
  if ((int)blockIdx.x >= *n_tiles) return;
  __shared__ __bf16 sA[2][16 * CHUNK];
  __shared__ int   s_tok[16];
  __shared__ float s_w[16];
  int tid  = threadIdx.x;
  int wave = tid >> 5;
  int lane = tid & 31;
  int lh   = lane >> 4;
  int nn   = lane & 15;
  int e    = tile_e[blockIdx.x];
  int row0 = tile_r0[blockIdx.x];
  if (tid < 16) { s_tok[tid] = perm_token[row0 + tid]; s_w[tid] = perm_rw[row0 + tid]; }

  int ncol0 = blockIdx.y * 512 + wave * 64;
  const __bf16* wbase = Wd + (size_t)e * H_ * I_;
  const __bf16* bC[4];
#pragma unroll
  for (int j = 0; j < 4; ++j)
    bC[j] = wbase + (size_t)(ncol0 + 16 * j + nn) * I_ + 8 * lh;

  v8f acc[4] = {{}, {}, {}, {}};

  int sm  = tid >> 4;
  int sc8 = (tid & 15) << 3;
  unsigned gl_off = (unsigned)(((row0 + sm) * I_ + sc8) * sizeof(__bf16));
  unsigned lds0 = (unsigned)(uintptr_t)&sA[0][sm * CHUNK + sc8];
  unsigned lds1 = (unsigned)(uintptr_t)&sA[1][sm * CHUNK + sc8];
  unsigned long long abase = (unsigned long long)(uintptr_t)act;

  async_copy_b128(lds0, gl_off, abase);
  async_wait0();
  __syncthreads();

  int p = 0;
  for (int kc = 0; kc < I_; kc += CHUNK, p ^= 1) {
    if (kc + CHUNK < I_)
      async_copy_b128(p ? lds0 : lds1,
                      gl_off + (unsigned)((kc + CHUNK) * sizeof(__bf16)), abase);
#pragma unroll
    for (int kk = 0; kk < CHUNK; kk += 32) {
      v16bf a = load_frag(&sA[p][nn * CHUNK + kk + 8 * lh]);
#pragma unroll
      for (int j = 0; j < 4; ++j) {
        v16bf b = load_frag(bC[j] + kc + kk);
        acc[j] = WMMA_BF16(a, b, acc[j]);
      }
    }
    async_wait0();
    __syncthreads();
  }

#pragma unroll
  for (int j = 0; j < 4; ++j) {
    int col = ncol0 + 16 * j + nn;
#pragma unroll
    for (int r = 0; r < 8; ++r) {
      int m = r + 8 * lh;
      int t = s_tok[m];
      if (t >= 0) unsafeAtomicAdd(&out[(size_t)t * H_ + col], s_w[m] * acc[j][r]);
    }
  }
}

// ------------------------------------------------------------------ launch
extern "C" void kernel_launch(void* const* d_in, const int* in_sizes, int n_in,
                              void* d_out, int out_size, void* d_ws, size_t ws_size,
                              hipStream_t stream) {
  const float* hidden = (const float*)d_in[0];   // [T,H]
  const float* rw     = (const float*)d_in[1];   // [T,K]
  const float* gup    = (const float*)d_in[2];   // [E,H,2I]
  const float* dwn    = (const float*)d_in[3];   // [E,I,H]
  const int*   sel    = (const int*)d_in[4];     // [T,K]
  float* out = (float*)d_out;

  char* ws = (char*)d_ws;
  size_t off = 0;
  auto take = [&](size_t bytes) -> char* {
    char* p = ws + off;
    off = (off + bytes + 255) & ~(size_t)255;
    return p;
  };
  __bf16* Wg   = (__bf16*)take((size_t)E_ * I2 * H_ * 2);   // 50.3 MB [E][2I][H]
  __bf16* Wd   = (__bf16*)take((size_t)E_ * H_ * I_ * 2);   // 25.2 MB [E][H][I]
  __bf16* Xp   = (__bf16*)take((size_t)PADROWS * H_ * 2);   //  8.9 MB
  __bf16* actb = (__bf16*)take((size_t)PADROWS * I_ * 2);   //  3.3 MB
  int*   perm_token = (int*)  take(PADROWS * 4);
  float* perm_rw    = (float*)take(PADROWS * 4);
  int*   tile_e     = (int*)  take(MAX_TILES * 4);
  int*   tile_r0    = (int*)  take(MAX_TILES * 4);
  int*   n_tiles    = (int*)  take(4);

  route_kernel<<<1, 256, 0, stream>>>(sel, rw, perm_token, perm_rw, tile_e, tile_r0, n_tiles);
  permute_cast_kernel<<<(PADROWS * H_ + 255) / 256, 256, 0, stream>>>(hidden, perm_token, Xp);

  dim3 tb(32, 8);
  transpose_cast_kernel<<<dim3(I2 / 32, H_ / 32, E_), tb, 0, stream>>>(gup, Wg, H_, I2);
  transpose_cast_kernel<<<dim3(H_ / 32, I_ / 32, E_), tb, 0, stream>>>(dwn, Wd, I_, H_);

  zero_out_kernel<<<(T_ * H_ + 255) / 256, 256, 0, stream>>>(out, T_ * H_);

  gemm1_swiglu_kernel<<<dim3(MAX_TILES, 2), 256, 0, stream>>>(Xp, Wg, tile_e, tile_r0,
                                                              n_tiles, actb);
  gemm2_combine_kernel<<<dim3(MAX_TILES, 4), 256, 0, stream>>>(actb, Wd, tile_e, tile_r0,
                                                               n_tiles, perm_token,
                                                               perm_rw, out);
}
